// ModuleCorrelation_1726576855364
// MI455X (gfx1250) — matrix operand order repaired and checked by reading, not compiled, not run on hardware
//
#include <hip/hip_runtime.h>

// CDNA5 / gfx1250 wave32 WMMA correlation kernel (banded-GEMM formulation).
typedef __bf16  v16bf __attribute__((ext_vector_type(16)));
typedef float   v8f   __attribute__((ext_vector_type(8)));

#define BATCH 8
#define CH    128
#define H     192
#define W     256
#define MD    3            // max displacement
#define ND    7            // 7x7 displacements
#define XT    64           // output pixels per block (x)
#define YT    16           // output rows per block (y)
#define SPX   80           // S pixels per LDS row: covers x0-3 .. x0+76
#define PITCH 136          // halves per pixel (128 ch + 8 pad) = 272 B, 16B-aligned rows
#define NSLOT 7            // ring buffer of 7 "second" rows
#define HW    (H * W)

// staging buffer: [dk (49)][x (64)] floats, pitch 66 words (conflict-free column writes)
#define SPITCH   66
#define SDUMP    (ND * ND * SPITCH)              // 3234 words: per-lane dump slots for invalid band entries
#define SWORDS   (SDUMP + 6 * ND * SPITCH + 64)  // + di*462 reach from dump + 32 lanes pad = 6070 words
#define LDSHALF  ((NSLOT * SPX + XT) * PITCH)    // 84,864 halves = 169,728 B (S ring + F row)

struct alignas(8) Bf4 { __bf16 a, b, c, d; };

__global__ __launch_bounds__(256)
void corr_wmma_kernel(const float* __restrict__ first,
                      const float* __restrict__ second,
                      float* __restrict__ out) {
    extern __shared__ unsigned short lds[];
    float* const stage = reinterpret_cast<float*>(lds + LDSHALF);

    const int tid = threadIdx.x;
    const int b   = blockIdx.z;
    const int y0  = blockIdx.y * YT;
    const int x0  = blockIdx.x * XT;

    const size_t planeHW = (size_t)HW;
    const float* Fb = first  + (size_t)b * CH * planeHW;
    const float* Sb = second + (size_t)b * CH * planeHW;

    const int FBASE = NSLOT * SPX * PITCH;       // F region after S ring buffer

    // shift/and fill decomposition: 16 lanes of x  x  16 channel-quads
    const int xi = tid & 15;
    const int ci = tid >> 4;                     // 0..15

    // ---- "second" row filler: 4 channels packed per ds_store_b64, native bf16 cvt ----
    auto loadS = [&](int g, int slot) {
        const bool rowOK = ((unsigned)g < (unsigned)H);
        const int  gc    = rowOK ? g : 0;
        #pragma unroll
        for (int cb = 0; cb < 2; ++cb) {
            const int    c0 = (cb * 16 + ci) * 4;
            const float* sp = Sb + (size_t)c0 * planeHW + (size_t)gc * W;
            #pragma unroll
            for (int pb = 0; pb < 5; ++pb) {
                const int px = pb * 16 + xi;                 // 0..79
                const int gx = x0 - MD + px;
                float v0 = 0.f, v1 = 0.f, v2 = 0.f, v3 = 0.f;
                if (rowOK && gx >= 0 && gx < W) {
                    v0 = sp[gx];
                    v1 = sp[planeHW + gx];
                    v2 = sp[2 * planeHW + gx];
                    v3 = sp[3 * planeHW + gx];
                }
                Bf4 pk = { (__bf16)v0, (__bf16)v1, (__bf16)v2, (__bf16)v3 };
                *reinterpret_cast<Bf4*>(&lds[(slot * SPX + px) * PITCH + c0]) = pk;
            }
        }
    };

    // ---- "first" row filler: 1/CH mean folded into the conversion ----
    auto loadF = [&](int y) {
        const float s = 1.f / CH;
        #pragma unroll
        for (int cb = 0; cb < 2; ++cb) {
            const int    c0 = (cb * 16 + ci) * 4;
            const float* fp = Fb + (size_t)c0 * planeHW + (size_t)y * W + x0;
            #pragma unroll
            for (int pb = 0; pb < 4; ++pb) {
                const int px = pb * 16 + xi;                 // 0..63
                Bf4 pk = { (__bf16)(fp[px] * s),
                           (__bf16)(fp[planeHW + px] * s),
                           (__bf16)(fp[2 * planeHW + px] * s),
                           (__bf16)(fp[3 * planeHW + px] * s) };
                *reinterpret_cast<Bf4*>(&lds[FBASE + px * PITCH + c0]) = pk;
            }
        }
    };

    const int lane   = tid & 31;
    const int wave   = tid >> 5;
    const int mt     = wave >> 1;                // 0..3 : 16-pixel output M-tile
    const int btile  = wave & 1;                 // 0/1  : shifted B tile
    const int lane16 = lane & 15;
    const int hi     = lane >> 4;

    // ---- per-lane staging word indices (invariant): invalid lanes -> dump slots ----
    // C/D layout: VGPR r, lane -> M = r + 8*hi, N = lane16 ; du = dj+3 = N - M + 16*btile
    int sidx[8];
    #pragma unroll
    for (int r = 0; r < 8; ++r) {
        const int M  = r + hi * 8;
        const int du = lane16 - M + btile * 16;
        const bool ok = (du >= 0) && (du < ND);
        sidx[r] = ok ? (du * SPITCH + mt * 16 + M) : (SDUMP + lane);
    }
    float* const outB = out + (size_t)b * (ND * ND) * HW;

    // ---- preload 7 "second" rows (y0-3..y0+3) + F row y0 ; incremental ring slots ----
    int sl = (y0 + 67) % 7;                      // slot of row y0-3 (single division, once)
    int sb = sl;                                 // slot of row (y-3) for the compute loop
    #pragma unroll 1
    for (int g = y0 - MD; g <= y0 + MD; ++g) {   // 7 rows; sl returns to slot(y0+4) == start
        loadS(g, sl);
        sl = (sl == 6) ? 0 : sl + 1;
    }
    loadF(y0);
    __syncthreads();

    for (int yy = 0; yy < YT; ++yy) {
        const int y = y0 + yy;

        // ---- A tiles (16x32 bf16, 4 K-chunks), kept in VGPRs across the di loop ----
        BfVec_decl:;
        union BfVec { v16bf v; uint4 q[2]; };
        BfVec a[4];
        {
            const unsigned short* frow = &lds[FBASE + (mt * 16 + lane16) * PITCH];
            #pragma unroll
            for (int kc = 0; kc < 4; ++kc) {
                const int k0 = kc * 32 + hi * 8;
                a[kc].q[0] = *reinterpret_cast<const uint4*>(frow + k0);
                a[kc].q[1] = *reinterpret_cast<const uint4*>(frow + k0 + 16);
            }
        }

        #pragma unroll
        for (int di = 0; di < ND; ++di) {
            int slot = sb + di;  if (slot >= NSLOT) slot -= NSLOT;
            const int spx = mt * 16 + btile * 16 + lane16;          // local S x index (0..79)
            const unsigned short* srow = &lds[(slot * SPX + spx) * PITCH];

            v8f acc = {0.f, 0.f, 0.f, 0.f, 0.f, 0.f, 0.f, 0.f};
            #pragma unroll
            for (int kc = 0; kc < 4; ++kc) {
                BfVec bv;
                const int k0 = kc * 32 + hi * 8;
                bv.q[0] = *reinterpret_cast<const uint4*>(srow + k0);
                bv.q[1] = *reinterpret_cast<const uint4*>(srow + k0 + 16);
                acc = __builtin_amdgcn_wmma_f32_16x16x32_bf16(
                          false, a[kc].v, false, bv.v, (short)0, acc, false, false);
            }

            // unconditional band staging: ds_store_b32 with immediate +di*462*4
            #pragma unroll
            for (int r = 0; r < 8; ++r)
                stage[(unsigned)sidx[r] + di * (ND * SPITCH)] = acc[r];
        }
        sb = (sb == 6) ? 0 : sb + 1;
        __syncthreads();

        // ---- overlap: fill next row's LDS (S/F regions) while writing out staging ----
        if (yy + 1 < YT) {
            loadS(y + MD + 1, sl);               // slot(y+4): overwrites slot(y-3), already consumed
            sl = (sl == 6) ? 0 : sl + 1;
            loadF(y + 1);
            // speculative prefetch of the row after that (global_prefetch_b8)
            const int c  = tid & 127;
            const int gy = y + MD + 2;
            if (gy < H) {
                int gx = x0 - MD + (tid >> 7) * 40;
                if (gx < 0) gx = 0;
                __builtin_prefetch(&Sb[(size_t)c * planeHW + (size_t)gy * W + gx], 0, 3);
            }
        }

        // ---- coalesced write-out: 49 dk rows x 64 x, contiguous 128B per wave ----
        {
            float* const outY = outB + (size_t)y * W + x0;
            int gidx = (tid >> 6) * HW + (tid & 63);           // dk0*HW + xl  (32-bit voffset)
            const int swrd = (tid >> 6) * SPITCH + (tid & 63); // staging word
            #pragma unroll
            for (int p = 0; p < 13; ++p) {                     // ceil(49*64/256)
                if (tid + 256 * p < ND * ND * XT)
                    outY[gidx] = stage[swrd + p * 4 * SPITCH]; // ds offset immediate
                gidx += 4 * HW;                                // dk advances by 4 per pass
            }
        }
        __syncthreads();
    }
}

extern "C" void kernel_launch(void* const* d_in, const int* in_sizes, int n_in,
                              void* d_out, int out_size, void* d_ws, size_t ws_size,
                              hipStream_t stream) {
    (void)in_sizes; (void)n_in; (void)out_size; (void)d_ws; (void)ws_size;
    const float* first  = (const float*)d_in[0];
    const float* second = (const float*)d_in[1];
    // d_in[2] = intStride (== 1 for this problem; kernel specialized for stride 1)
    float* out = (float*)d_out;

    dim3 grid(W / XT, H / YT, BATCH);   // (4, 12, 8)
    dim3 block(256);                    // 8 wave32s
    const size_t shmem = (size_t)LDSHALF * 2 + (size_t)SWORDS * 4;   // 169,728 + 24,280 B
    hipLaunchKernelGGL(corr_wmma_kernel, grid, block, shmem, stream, first, second, out);
}
